// LaneAllocationGAT_26216480375048
// MI455X (gfx1250) — matrix operand order repaired and checked by reading, not compiled, not run on hardware
//
#include <hip/hip_runtime.h>
#include <stdint.h>

#define N_NODES 50000
#define E_EDGES 800000
#define ETOT (E_EDGES + N_NODES)   // self loops appended
#define HC 128
#define HEADS 2
#define DOUT 4
#define NEG_SLOPE 0.2f
#define GEMM_ROWS 80               // 50000 = 625 * 80, exact

typedef __attribute__((ext_vector_type(16))) __bf16 v16bf;
typedef __attribute__((ext_vector_type(16))) unsigned short v16u;
typedef __attribute__((ext_vector_type(8)))  float v8f;
typedef __attribute__((ext_vector_type(4)))  unsigned int su4;
typedef __attribute__((ext_vector_type(8)))  unsigned int su8;

__device__ __forceinline__ unsigned short f32_to_bf16u(float f) {
  unsigned int u = __float_as_uint(f);
  return (unsigned short)((u + 0x7FFFu + ((u >> 16) & 1u)) >> 16);
}

__device__ __forceinline__ void atomicMaxFloat(float* addr, float v) {
  // sign-aware int/uint trick; init value is -1e30f (negative)
  if (v >= 0.f) atomicMax((int*)addr, __float_as_int(v));
  else          atomicMin((unsigned int*)addr, (unsigned int)__float_as_int(v));
}

__device__ __forceinline__ float lrelu(float v) {
  return v > 0.f ? v : NEG_SLOPE * v;
}

// ---------------------------------------------------------------- TDM 2D tile load
// Issue a Tensor Data Mover load of a (tile_d1 x 128) row-major bf16 tile
// (row stride = 128 elements) from global memory into LDS at byte offset lds_off.
// 2-group descriptor (VADDR2/3 = NULL => tensors up to 2D). Tracked by TENSORcnt.
__device__ __forceinline__ void tdm_load_tile_2d(const void* gaddr, unsigned lds_off,
                                                 unsigned tensor_d1, unsigned tile_d1) {
  // --- D# group 0 (128b): count=1 | lds_addr<<32 | global_addr | type=2<<126
  unsigned long long va = (unsigned long long)gaddr;
  unsigned long long p0 = 1ULL | ((unsigned long long)lds_off << 32);
  unsigned long long p1 = (va & 0x01FFFFFFFFFFFFFFULL) | (2ULL << 62);
  su4 g0 = { (unsigned)p0, (unsigned)(p0 >> 32), (unsigned)p1, (unsigned)(p1 >> 32) };
  // --- D# group 1 (256b): data_size=1(2B); tensor_dim0=128; tensor_dim1; tile_dim0=128;
  //     tile_dim1; tile_dim2=0; tensor_dim0_stride=128; tensor_dim1_stride=0; no flags.
  unsigned long long q0 = (1ULL << 16) | (128ULL << 48);                      // ds | td0.lo
  unsigned long long q1 = ((unsigned long long)tensor_d1 << 16) | (128ULL << 48); // td0.hi|td1|tile0
  unsigned long long q2 = (unsigned long long)tile_d1 | (128ULL << 32);       // tile1|tile2|str0.lo
  unsigned long long q3 = 0;                                                  // str0.hi|str1
  su8 g1 = { (unsigned)q0, (unsigned)(q0 >> 32), (unsigned)q1, (unsigned)(q1 >> 32),
             (unsigned)q2, (unsigned)(q2 >> 32), (unsigned)q3, (unsigned)(q3 >> 32) };
  asm volatile("tensor_load_to_lds %0, %1" :: "s"(g0), "s"(g1) : "memory");
}

// ---------------------------------------------------------------- convert
__global__ void k_f32_to_bf16(const float* __restrict__ in,
                              unsigned short* __restrict__ out, int n) {
  int i = blockIdx.x * blockDim.x + threadIdx.x;
  if (i < n) out[i] = f32_to_bf16u(in[i]);
}

// ---------------------------------------------------------------- WMMA GEMM
// Y[M x 128] = bf16(X[M x 128]) @ bf16(W[128 x 128]) + bias
// 256 threads = 8 waves; wave w owns column tile n0 = 16*w; 5 row tiles of 16.
// A strip + W staged into LDS by the Tensor Data Mover (wave 0 issues the DMA).
__global__ void __launch_bounds__(256) k_gemm_bias(
    const unsigned short* __restrict__ X, const unsigned short* __restrict__ W,
    const float* __restrict__ bias, float* __restrict__ Y) {
  extern __shared__ unsigned short smem[];                 // dynamic LDS, base offset 0
  unsigned short* sA = smem;                               // [GEMM_ROWS*HC] @ byte 0
  unsigned short* sW = smem + GEMM_ROWS * HC;              // [HC*HC]       @ byte 20480
  const int tid  = threadIdx.x;
  const int wave = tid >> 5;
  const int lane = tid & 31;
  const int row0 = blockIdx.x * GEMM_ROWS;

  if (tid < 32) {  // one wave drives the TDM; EXEC is ignored by tensor ops
    tdm_load_tile_2d(X + (size_t)row0 * HC, 0u,            N_NODES, GEMM_ROWS);
    tdm_load_tile_2d(W,                     GEMM_ROWS * HC * 2u, HC, HC);
    __builtin_amdgcn_s_wait_tensorcnt(0);
  }
  __syncthreads();

  union BU { v16u v; uint4 q[2]; };

  // B fragments: lane L holds column n0 + (L&15), K = {k0+8h..k0+8h+7, k0+16+8h..k0+16+8h+7}, h = L>>4.
  const int n0 = wave * 16;
  BU bfrag[4];
#pragma unroll
  for (int kk = 0; kk < 4; ++kk) {
#pragma unroll
    for (int s = 0; s < 16; ++s) {
      const int kb = kk * 32 + (s < 8 ? 8 * (lane >> 4) + s : 16 + 8 * (lane >> 4) + (s - 8));
      bfrag[kk].v[s] = sW[kb * HC + n0 + (lane & 15)];
    }
  }

  const int m = lane & 15;   // A row within tile / D column
  const int g = lane >> 4;   // half-wave selector

#pragma unroll
  for (int rt = 0; rt < GEMM_ROWS / 16; ++rt) {
    v8f c = {};
    const unsigned short* arow = sA + (rt * 16 + m) * HC;
#pragma unroll
    for (int kk = 0; kk < 4; ++kk) {
      const int k0 = kk * 32;
      BU a;
      // 16-bit A layout: lane(m,g) holds K = {k0+8g..k0+8g+7, k0+16+8g..k0+16+8g+7}
      a.q[0] = *(const uint4*)(arow + k0 + 8 * g);
      a.q[1] = *(const uint4*)(arow + k0 + 16 + 8 * g);
      c = __builtin_amdgcn_wmma_f32_16x16x32_bf16(
              false, __builtin_bit_cast(v16bf, a.v),
              false, __builtin_bit_cast(v16bf, bfrag[kk].v),
              (short)0, c, false, false);
    }
    const int col = n0 + m;
    const float bv = bias[col];
#pragma unroll
    for (int r = 0; r < 8; ++r) {
      // D layout: VGPR r -> row (r + 8*g), col = lane&15
      Y[(size_t)(row0 + rt * 16 + 8 * g + r) * HC + col] = c[r] + bv;
    }
  }
}

// ---------------------------------------------------------------- init buffers
__global__ void k_init(float* __restrict__ emax, float* __restrict__ den,
                       float* __restrict__ agg) {
  int i = blockIdx.x * blockDim.x + threadIdx.x;
  if (i < N_NODES * HEADS) { emax[i] = -1e30f; den[i] = 0.f; }
  if (i < N_NODES * HC) agg[i] = 0.f;
}

// ---------------------------------------------------------------- edge scores + segment max
__global__ void __launch_bounds__(256) k_edge_scores(
    const long long* __restrict__ srcp, const long long* __restrict__ dstp,
    const float* __restrict__ xl, const float* __restrict__ xr,
    const float* __restrict__ att, float* __restrict__ esc,
    float* __restrict__ emax) {
  const int lane = threadIdx.x & 31;
  const long long eid = (long long)blockIdx.x * 8 + (threadIdx.x >> 5);
  if (eid >= ETOT) return;
  long long s, d;
  if (eid < E_EDGES) { s = srcp[eid]; d = dstp[eid]; }
  else               { s = d = eid - E_EDGES; }

  float4 a = *(const float4*)(xl + s * HC + lane * 4);
  float4 b = *(const float4*)(xr + d * HC + lane * 4);
  float4 t = ((const float4*)att)[lane];   // att flat [H*C] matches feature index

  float acc = lrelu(a.x + b.x) * t.x + lrelu(a.y + b.y) * t.y +
              lrelu(a.z + b.z) * t.z + lrelu(a.w + b.w) * t.w;
  // head 0 = lanes 0..15, head 1 = lanes 16..31
#pragma unroll
  for (int off = 8; off; off >>= 1) acc += __shfl_xor(acc, off, 16);
  if ((lane & 15) == 0) {
    const int h = lane >> 4;
    esc[eid * HEADS + h] = acc;
    atomicMaxFloat(&emax[d * HEADS + h], acc);
  }
}

// ---------------------------------------------------------------- exp + segment sum
__global__ void k_edge_expsum(const long long* __restrict__ dstp,
                              const float* __restrict__ emax,
                              float* __restrict__ esc,
                              float* __restrict__ den) {
  long long i = (long long)blockIdx.x * blockDim.x + threadIdx.x;
  if (i >= (long long)ETOT * HEADS) return;
  const long long eid = i >> 1;
  const int h = (int)(i & 1);
  const long long d = (eid < E_EDGES) ? dstp[eid] : (eid - E_EDGES);
  const float p = __expf(esc[i] - emax[d * HEADS + h]);
  esc[i] = p;
  atomicAdd(&den[d * HEADS + h], p);
}

// ---------------------------------------------------------------- weighted scatter-add
__global__ void __launch_bounds__(256) k_edge_aggregate(
    const long long* __restrict__ srcp, const long long* __restrict__ dstp,
    const float* __restrict__ xl, const float* __restrict__ esc,
    const float* __restrict__ den, float* __restrict__ agg) {
  const int lane = threadIdx.x & 31;
  const long long eid = (long long)blockIdx.x * 8 + (threadIdx.x >> 5);
  if (eid >= ETOT) return;
  long long s, d;
  if (eid < E_EDGES) { s = srcp[eid]; d = dstp[eid]; }
  else               { s = d = eid - E_EDGES; }
  const int h = lane >> 4;
  const float alpha = esc[eid * HEADS + h] / den[d * HEADS + h];
  float4 a = *(const float4*)(xl + s * HC + lane * 4);
  float* base = agg + d * HC + lane * 4;
  atomicAdd(base + 0, alpha * a.x);
  atomicAdd(base + 1, alpha * a.y);
  atomicAdd(base + 2, alpha * a.z);
  atomicAdd(base + 3, alpha * a.w);
}

// ---------------------------------------------------------------- bias + relu + bf16 copy
__global__ void k_finalize(float* __restrict__ agg, const float* __restrict__ bias,
                           unsigned short* __restrict__ xbf) {
  int i = blockIdx.x * blockDim.x + threadIdx.x;
  if (i >= N_NODES * HC) return;
  float v = agg[i] + bias[i & (HC - 1)];
  v = v > 0.f ? v : 0.f;
  agg[i] = v;
  xbf[i] = f32_to_bf16u(v);
}

// ---------------------------------------------------------------- output linear (128 -> 4)
__global__ void k_out(const float* __restrict__ h, const float* __restrict__ Wout,
                      const float* __restrict__ bout, float* __restrict__ out) {
  int i = blockIdx.x * blockDim.x + threadIdx.x;
  if (i >= N_NODES * DOUT) return;
  const int n = i >> 2, j = i & 3;
  float acc = bout[j];
  const float* hp = h + (size_t)n * HC;
#pragma unroll 8
  for (int k = 0; k < HC; ++k) acc += hp[k] * Wout[k * DOUT + j];
  out[i] = acc;
}

// ----------------------------------------------------------------------------
extern "C" void kernel_launch(void* const* d_in, const int* in_sizes, int n_in,
                              void* d_out, int out_size, void* d_ws, size_t ws_size,
                              hipStream_t stream) {
  (void)in_sizes; (void)n_in; (void)out_size; (void)ws_size;
  const float*     x    = (const float*)d_in[0];
  const long long* ei   = (const long long*)d_in[1];
  const long long* srcp = ei;
  const long long* dstp = ei + E_EDGES;
  const float* Wl[3]   = {(const float*)d_in[2],  (const float*)d_in[8],  (const float*)d_in[14]};
  const float* bl[3]   = {(const float*)d_in[3],  (const float*)d_in[9],  (const float*)d_in[15]};
  const float* Wr[3]   = {(const float*)d_in[4],  (const float*)d_in[10], (const float*)d_in[16]};
  const float* br[3]   = {(const float*)d_in[5],  (const float*)d_in[11], (const float*)d_in[17]};
  const float* att[3]  = {(const float*)d_in[6],  (const float*)d_in[12], (const float*)d_in[18]};
  const float* bias[3] = {(const float*)d_in[7],  (const float*)d_in[13], (const float*)d_in[19]};
  const float* Wout = (const float*)d_in[20];
  const float* bout = (const float*)d_in[21];

  char* ws = (char*)d_ws;
  size_t off = 0;
  auto take = [&](size_t bytes) -> void* {
    void* p = ws + off;
    off += (bytes + 255) & ~(size_t)255;
    return p;
  };
  unsigned short* xbf  = (unsigned short*)take((size_t)N_NODES * HC * 2);
  unsigned short* wlbf = (unsigned short*)take((size_t)HC * HC * 2);
  unsigned short* wrbf = (unsigned short*)take((size_t)HC * HC * 2);
  float* xl   = (float*)take((size_t)N_NODES * HC * 4);
  float* xr   = (float*)take((size_t)N_NODES * HC * 4);
  float* esc  = (float*)take((size_t)ETOT * HEADS * 4);
  float* emax = (float*)take((size_t)N_NODES * HEADS * 4);
  float* den  = (float*)take((size_t)N_NODES * HEADS * 4);
  float* agg  = (float*)take((size_t)N_NODES * HC * 4);

  const int elemNHC  = N_NODES * HC;            // 6.4M
  const int gridNHC  = (elemNHC + 255) / 256;   // 25000
  const int gridEdge = ETOT / 8;                // 106250 exact
  const int gridEH   = (ETOT * HEADS + 255) / 256;
  const size_t gemmLds = (size_t)(GEMM_ROWS * HC + HC * HC) * 2;  // 53248 B

  k_f32_to_bf16<<<gridNHC, 256, 0, stream>>>(x, xbf, elemNHC);

  for (int L = 0; L < 3; ++L) {
    k_f32_to_bf16<<<(HC * HC + 255) / 256, 256, 0, stream>>>(Wl[L], wlbf, HC * HC);
    k_f32_to_bf16<<<(HC * HC + 255) / 256, 256, 0, stream>>>(Wr[L], wrbf, HC * HC);
    k_gemm_bias<<<N_NODES / GEMM_ROWS, 256, gemmLds, stream>>>(xbf, wlbf, bl[L], xl);
    k_gemm_bias<<<N_NODES / GEMM_ROWS, 256, gemmLds, stream>>>(xbf, wrbf, br[L], xr);
    k_init<<<gridNHC, 256, 0, stream>>>(emax, den, agg);
    k_edge_scores<<<gridEdge, 256, 0, stream>>>(srcp, dstp, xl, xr, att[L], esc, emax);
    k_edge_expsum<<<gridEH, 256, 0, stream>>>(dstp, emax, esc, den);
    k_edge_aggregate<<<gridEdge, 256, 0, stream>>>(srcp, dstp, xl, esc, den, agg);
    k_finalize<<<gridNHC, 256, 0, stream>>>(agg, bias[L], xbf);
  }
  k_out<<<(N_NODES * DOUT + 255) / 256, 256, 0, stream>>>(agg, Wout, bout, (float*)d_out);
}
